// RAFT_52012053955260
// MI455X (gfx1250) — matrix operand order repaired and checked by reading, not compiled, not run on hardware
//
#include <hip/hip_runtime.h>
#include <hip/hip_bf16.h>
#include <cstdint>
#include <cstddef>

// ---------------------------------------------------------------------------
// RAFT forward for MI455X (gfx1250).  All conv / correlation GEMMs run on
// V_WMMA_F32_16X16X32_F16 (f16 in, f32 accumulate), 32x32 output tile per
// wave (4 WMMA per K-chunk).  Correlation GEMM uses double-buffered
// global_load_async_to_lds_b128 staging when the toolchain exposes it.
// ---------------------------------------------------------------------------

typedef __attribute__((ext_vector_type(16))) _Float16 v16h;
typedef __attribute__((ext_vector_type(8)))  float    v8f;
typedef __attribute__((ext_vector_type(4)))  int      v4i;

#define HF   46
#define WF   96
#define HWF  (HF * WF)          // 4416
#define HIN  368
#define WIN  768

#if defined(__HIP_DEVICE_COMPILE__) && __has_builtin(__builtin_amdgcn_global_load_async_to_lds_b128)
#define HAS_ASYNC_LDS 1
typedef __attribute__((address_space(1))) v4i GV4;   // global int4
typedef __attribute__((address_space(3))) v4i LV4;   // LDS int4
static __device__ __forceinline__ void async_g2l_b128(const void* g, void* l) {
    __builtin_amdgcn_global_load_async_to_lds_b128((GV4*)g, (LV4*)l, 0, 0);
}
#if __has_builtin(__builtin_amdgcn_s_wait_asynccnt)
#define WAIT_ASYNC(n) __builtin_amdgcn_s_wait_asynccnt(n)
#else
#define WAIT_ASYNC(n) asm volatile("s_wait_asynccnt %0" ::"i"(n) : "memory")
#endif
#endif

static __device__ __forceinline__ float sigmoidf_(float x) {
    return 1.0f / (1.0f + __expf(-x));
}

// ------------------------- generic conv as implicit GEMM -------------------
// One wave computes a 32(M) x 32(N) output tile; M = Nimg*Hout*Wout,
// K = Cin*KH*KW.  A = im2col(input) f32->f16 on the fly, B = weights^T.
// Row/K descriptors are precomputed into LDS once per block / per chunk.
__global__ __launch_bounds__(32)
void k_conv_wmma(const float* __restrict__ in, const float* __restrict__ w,
                 const float* __restrict__ bias, float* __restrict__ out,
                 int Nimg, int Cin, int Hin, int Win,
                 int Cout, int KH, int KW, int stride, int padH, int padW,
                 int Hout, int Wout, int act, float oscale)
{
    const int M  = Nimg * Hout * Wout;
    const int K  = Cin * KH * KW;
    const int tn_count = (Cout + 31) >> 5;
    const int tn = blockIdx.x % tn_count;
    const int tm = blockIdx.x / tn_count;
    const int m0 = tm << 5, n0 = tn << 5;
    const int lane = threadIdx.x;            // 0..31 (wave32)
    const int HWout = Hout * Wout;
    const int KHW = KH * KW;

    __shared__ _Float16 As[32 * 32];         // [m][k]
    __shared__ _Float16 Bs[32 * 32];         // [k][n]
    __shared__ int rImg[32], rIY[32], rIX[32];
    __shared__ int kCI[32], kKY[32], kKX[32];

    // per-block row descriptors (one row per lane)
    {
        int m = m0 + lane;
        int img = -1, oy = 0, ox = 0;
        if (m < M) {
            img = m / HWout;
            int rem = m - img * HWout;
            oy = rem / Wout; ox = rem - oy * Wout;
        }
        rImg[lane] = img;
        rIY[lane] = oy * stride - padH;
        rIX[lane] = ox * stride - padW;
    }

    v8f a00 = {}, a01 = {}, a10 = {}, a11 = {};

    for (int k0 = 0; k0 < K; k0 += 32) {
        if (k0 + 32 < K)                     // global_prefetch_b8 next W chunk
            __builtin_prefetch(&w[(size_t)n0 * K + k0 + 32], 0, 1);
        // per-chunk K descriptors (one k per lane)
        {
            int k = k0 + lane;
            int ci = -1, ky = 0, kx = 0;
            if (k < K) {
                ci = k / KHW;
                int kr = k - ci * KHW;
                ky = kr / KW; kx = kr - ky * KW;
            }
            kCI[lane] = ci; kKY[lane] = ky; kKX[lane] = kx;
        }
        __syncthreads();
        // A staging: lane owns one output row
        {
            const int img = rImg[lane];
            const int iyb = rIY[lane], ixb = rIX[lane];
            #pragma unroll 4
            for (int ki = 0; ki < 32; ++ki) {
                float av = 0.0f;
                int ci = kCI[ki];
                if (img >= 0 && ci >= 0) {
                    int iy = iyb + kKY[ki];
                    int ix = ixb + kKX[ki];
                    if (iy >= 0 && iy < Hin && ix >= 0 && ix < Win)
                        av = in[(((size_t)img * Cin + ci) * Hin + iy) * Win + ix];
                }
                As[lane * 32 + ki] = (_Float16)av;
            }
        }
        // B staging: lane owns one output channel, contiguous K read
        {
            int n = n0 + lane;
            bool nv = n < Cout;
            const float* wp = w + (size_t)n * K + k0;
            #pragma unroll 4
            for (int ki = 0; ki < 32; ++ki) {
                float bv = (nv && (k0 + ki) < K) ? wp[ki] : 0.0f;
                Bs[ki * 32 + lane] = (_Float16)bv;
            }
        }
        __syncthreads();
        // fragment assembly per CDNA5 16-bit layouts
        v16h a0, a1, b0, b1;
        const int hf = lane >> 4, mr = lane & 15, nc = lane & 15;
        #pragma unroll
        for (int j = 0; j < 8; ++j) {
            a0[j]     = As[mr * 32 + hf * 8 + j];
            a0[j + 8] = As[mr * 32 + 16 + hf * 8 + j];
            a1[j]     = As[(16 + mr) * 32 + hf * 8 + j];
            a1[j + 8] = As[(16 + mr) * 32 + 16 + hf * 8 + j];
        }
        #pragma unroll
        for (int j = 0; j < 16; ++j) {
            b0[j] = Bs[((j >> 3) * 16 + hf * 8 + (j & 7)) * 32 + nc];
            b1[j] = Bs[((j >> 3) * 16 + hf * 8 + (j & 7)) * 32 + 16 + nc];
        }
        a00 = __builtin_amdgcn_wmma_f32_16x16x32_f16(false, a0, false, b0, (short)0, a00, false, false);
        a01 = __builtin_amdgcn_wmma_f32_16x16x32_f16(false, a0, false, b1, (short)0, a01, false, false);
        a10 = __builtin_amdgcn_wmma_f32_16x16x32_f16(false, a1, false, b0, (short)0, a10, false, false);
        a11 = __builtin_amdgcn_wmma_f32_16x16x32_f16(false, a1, false, b1, (short)0, a11, false, false);
        __syncthreads();
    }

    // store 4 sub-tiles
    const v8f* accs[4] = { &a00, &a01, &a10, &a11 };
    #pragma unroll
    for (int s = 0; s < 4; ++s) {
        int ti = s >> 1, tj = s & 1;
        int n = n0 + tj * 16 + (lane & 15);
        if (n >= Cout) continue;
        const float bv = bias ? bias[n] : 0.0f;
        #pragma unroll
        for (int r = 0; r < 8; ++r) {
            int m = m0 + ti * 16 + r + 8 * (lane >> 4);
            if (m < M) {
                float v = ((*accs[s])[r] + bv) * oscale;
                if (act) v = fmaxf(v, 0.0f);
                int img = m / HWout;
                int rem = m - img * HWout;
                out[(((size_t)img * Cout + n) * Hout + rem / Wout) * Wout
                    + (rem % Wout)] = v;
            }
        }
    }
}

// -------- correlation GEMM ------------------------------------------------
// f1t, f2t are transposed f16 feature maps: [position][channel] (C=256).
// corr[m][n] = scale * sum_k f1t[m][k] * f2t[n][k].
// Async path: double-buffered global_load_async_to_lds_b128 staging.
__global__ __launch_bounds__(32)
void k_corr_wmma(const _Float16* __restrict__ f1, const _Float16* __restrict__ f2,
                 float* __restrict__ out, int C, int P, float scale)
{
    const int tiles = P >> 5;                // P multiple of 32 (4416/32 = 138)
    const int tn = blockIdx.x % tiles;
    const int tm = blockIdx.x / tiles;
    const int m0 = tm << 5, n0 = tn << 5;
    const int lane = threadIdx.x;

#ifdef HAS_ASYNC_LDS
    __shared__ _Float16 As[2][32 * 32];      // [m][k]
    __shared__ _Float16 Bs[2][32 * 32];      // [n][k]
#else
    __shared__ _Float16 As[1][32 * 32];
    __shared__ _Float16 Bs[1][32 * 32];
#endif
    v8f a00 = {}, a01 = {}, a10 = {}, a11 = {};
    const int nchunk = C >> 5;

#ifdef HAS_ASYNC_LDS
    // each stage = 8 async b128 copies (4 for A tile, 4 for B tile)
    auto stage = [&](int buf, int k0) {
        #pragma unroll
        for (int op = 0; op < 4; ++op) {
            int o   = (op * 32 + lane) * 16;        // byte offset in 2KB tile
            int row = o >> 6;                       // 64 bytes per row
            int col = o & 63;
            async_g2l_b128((const char*)f1 + ((size_t)(m0 + row) * C + k0) * 2 + col,
                           (char*)&As[buf][0] + o);
            async_g2l_b128((const char*)f2 + ((size_t)(n0 + row) * C + k0) * 2 + col,
                           (char*)&Bs[buf][0] + o);
        }
    };
    stage(0, 0);
#endif

    for (int t = 0; t < nchunk; ++t) {
        const int k0 = t << 5;
#ifdef HAS_ASYNC_LDS
        if (t + 1 < nchunk) {
            stage((t + 1) & 1, k0 + 32);
            WAIT_ASYNC(8);                   // current buffer's 8 ops complete
        } else {
            WAIT_ASYNC(0);
        }
        const _Float16* Ap = &As[t & 1][0];
        const _Float16* Bp = &Bs[t & 1][0];
#else
        #pragma unroll 4
        for (int j = 0; j < 32; ++j) {
            As[0][lane * 32 + j] = f1[(size_t)(m0 + lane) * C + k0 + j];
            Bs[0][lane * 32 + j] = f2[(size_t)(n0 + lane) * C + k0 + j];
        }
        __syncthreads();
        const _Float16* Ap = &As[0][0];
        const _Float16* Bp = &Bs[0][0];
#endif
        v16h a0, a1, b0, b1;
        const int hf = lane >> 4, mr = lane & 15, nc = lane & 15;
        #pragma unroll
        for (int j = 0; j < 8; ++j) {
            a0[j]     = Ap[mr * 32 + hf * 8 + j];
            a0[j + 8] = Ap[mr * 32 + 16 + hf * 8 + j];
            a1[j]     = Ap[(16 + mr) * 32 + hf * 8 + j];
            a1[j + 8] = Ap[(16 + mr) * 32 + 16 + hf * 8 + j];
        }
        #pragma unroll
        for (int j = 0; j < 16; ++j) {
            b0[j] = Bp[nc * 32 + (j >> 3) * 16 + hf * 8 + (j & 7)];          // B[k][n] from [n][k]
            b1[j] = Bp[(16 + nc) * 32 + (j >> 3) * 16 + hf * 8 + (j & 7)];
        }
        a00 = __builtin_amdgcn_wmma_f32_16x16x32_f16(false, a0, false, b0, (short)0, a00, false, false);
        a01 = __builtin_amdgcn_wmma_f32_16x16x32_f16(false, a0, false, b1, (short)0, a01, false, false);
        a10 = __builtin_amdgcn_wmma_f32_16x16x32_f16(false, a1, false, b0, (short)0, a10, false, false);
        a11 = __builtin_amdgcn_wmma_f32_16x16x32_f16(false, a1, false, b1, (short)0, a11, false, false);
#ifndef HAS_ASYNC_LDS
        __syncthreads();
#endif
    }

    const v8f* accs[4] = { &a00, &a01, &a10, &a11 };
    #pragma unroll
    for (int s = 0; s < 4; ++s) {
        int ti = s >> 1, tj = s & 1;
        int n = n0 + tj * 16 + (lane & 15);
        #pragma unroll
        for (int r = 0; r < 8; ++r) {
            int m = m0 + ti * 16 + r + 8 * (lane >> 4);
            out[(size_t)m * P + n] = (*accs[s])[r] * scale;
        }
    }
}

// --------------------------- per-channel-plane norm ------------------------
__global__ __launch_bounds__(256)
void k_norm(float* __restrict__ x, const float* __restrict__ gamma,
            const float* __restrict__ beta, int C, int HW, int relu)
{
    const int nc = blockIdx.x;
    float* p = x + (size_t)nc * HW;
    __shared__ float ss[256], sq[256];
    float s = 0.0f, q = 0.0f;
    for (int i = threadIdx.x; i < HW; i += 256) {
        float v = p[i]; s += v; q += v * v;
    }
    ss[threadIdx.x] = s; sq[threadIdx.x] = q;
    __syncthreads();
    for (int off = 128; off > 0; off >>= 1) {
        if (threadIdx.x < off) {
            ss[threadIdx.x] += ss[threadIdx.x + off];
            sq[threadIdx.x] += sq[threadIdx.x + off];
        }
        __syncthreads();
    }
    const float mean = ss[0] / (float)HW;
    const float var  = sq[0] / (float)HW - mean * mean;
    const float inv  = rsqrtf(var + 1e-5f);
    const int c = nc % C;
    const float g = gamma ? gamma[c] : 1.0f;
    const float b = beta  ? beta[c]  : 0.0f;
    for (int i = threadIdx.x; i < HW; i += 256) {
        float v = (p[i] - mean) * inv * g + b;
        if (relu) v = fmaxf(v, 0.0f);
        p[i] = v;
    }
}

// ------------------------------- pointwise ---------------------------------
__global__ void k_scale2m1(const float* a, float* o, int n) {
    int i = blockIdx.x * blockDim.x + threadIdx.x;
    if (i < n) o[i] = 2.0f * a[i] - 1.0f;
}
// transpose-convert: out[img][p][c] = (f16) in[img][c][p]
__global__ void k_cvt_t(const float* a, _Float16* o, int imgs) {
    int i = blockIdx.x * blockDim.x + threadIdx.x;
    int total = imgs * 256 * HWF;
    if (i >= total) return;
    int img = i / (256 * HWF);
    int r = i - img * (256 * HWF);
    int p = r >> 8;
    int c = r & 255;
    o[i] = (_Float16)a[((size_t)img * 256 + c) * HWF + p];
}
__global__ void k_add_relu(const float* a, const float* b, float* o, int n) {
    int i = blockIdx.x * blockDim.x + threadIdx.x;
    if (i < n) o[i] = fmaxf(a[i] + b[i], 0.0f);
}
__global__ void k_copy(const float* s, float* d, int n) {
    int i = blockIdx.x * blockDim.x + threadIdx.x;
    if (i < n) d[i] = s[i];
}
__global__ void k_add_ip(float* a, const float* b, int n) {
    int i = blockIdx.x * blockDim.x + threadIdx.x;
    if (i < n) a[i] += b[i];
}
__global__ void k_split_cnet(const float* c, float* net, float* hx, float* rhx, int n) {
    int i = blockIdx.x * blockDim.x + threadIdx.x;
    if (i < n) {
        net[i] = tanhf(c[i]);
        float v = fmaxf(c[n + i], 0.0f);
        hx[n + i] = v; rhx[n + i] = v;
    }
}
__global__ void k_init_coords(float* c) {
    int i = blockIdx.x * blockDim.x + threadIdx.x;
    if (i < HWF) { c[i] = (float)(i % WF); c[HWF + i] = (float)(i / WF); }
}
__global__ void k_flow(const float* c1, float* f) {
    int i = blockIdx.x * blockDim.x + threadIdx.x;
    if (i < HWF) {
        f[i]       = c1[i]       - (float)(i % WF);
        f[HWF + i] = c1[HWF + i] - (float)(i / WF);
    }
}
__global__ void k_gru_zr(const float* zp, const float* rp, const float* net,
                         float* z, float* rh, int n) {
    int i = blockIdx.x * blockDim.x + threadIdx.x;
    if (i < n) {
        z[i]  = sigmoidf_(zp[i]);
        rh[i] = sigmoidf_(rp[i]) * net[i];
    }
}
__global__ void k_gru_out(const float* qp, const float* z, float* net, int n) {
    int i = blockIdx.x * blockDim.x + threadIdx.x;
    if (i < n) {
        float zz = z[i];
        net[i] = (1.0f - zz) * net[i] + zz * tanhf(qp[i]);
    }
}
__global__ void k_avgpool(const float* in, float* out, int B, int hi, int wi) {
    int ho = hi >> 1, wo = wi >> 1;
    int total = B * ho * wo;
    int i = blockIdx.x * blockDim.x + threadIdx.x;
    if (i >= total) return;
    int b = i / (ho * wo);
    int r = i - b * (ho * wo);
    int y = r / wo, x = r - y * wo;
    const float* p = in + (size_t)b * hi * wi + (2 * y) * wi + 2 * x;
    out[i] = 0.25f * (p[0] + p[1] + p[wi] + p[wi + 1]);
}

static __device__ __forceinline__
float tap_(const float* img, int hh, int ww, int yy, int xx, float wgt) {
    if (xx < 0 || xx > ww - 1 || yy < 0 || yy > hh - 1) return 0.0f;
    return img[yy * ww + xx] * wgt;
}

__global__ void k_corr_lookup(const float* __restrict__ l0, const float* __restrict__ l1,
                              const float* __restrict__ l2, const float* __restrict__ l3,
                              const float* __restrict__ coords, float* __restrict__ out)
{
    int idx = blockIdx.x * blockDim.x + threadIdx.x;
    if (idx >= 324 * HWF) return;
    int p   = idx % HWF;
    int ch  = idx / HWF;
    int lev = ch / 81, t = ch % 81;
    const float* img; int hh, ww;
    if      (lev == 0) { img = l0; hh = 46; ww = 96; }
    else if (lev == 1) { img = l1; hh = 23; ww = 48; }
    else if (lev == 2) { img = l2; hh = 11; ww = 24; }
    else               { img = l3; hh = 5;  ww = 12; }
    img += (size_t)p * hh * ww;
    const float inv = 1.0f / (float)(1 << lev);
    // RAFT quirk: first meshgrid axis is added to x
    float x = coords[p] * inv       + (float)(t / 9 - 4);
    float y = coords[HWF + p] * inv + (float)(t % 9 - 4);
    float x0f = floorf(x), y0f = floorf(y);
    float tx = x - x0f, ty = y - y0f;
    int x0 = (int)x0f, y0 = (int)y0f;
    float acc = tap_(img, hh, ww, y0,     x0,     (1.0f - tx) * (1.0f - ty))
              + tap_(img, hh, ww, y0,     x0 + 1, tx * (1.0f - ty))
              + tap_(img, hh, ww, y0 + 1, x0,     (1.0f - tx) * ty)
              + tap_(img, hh, ww, y0 + 1, x0 + 1, tx * ty);
    out[(size_t)ch * HWF + p] = acc;
}

// convex upsample: softmax over 9 mask weights, combine 3x3 patch of 8*flow
__global__ void k_upsample(const float* __restrict__ flow, const float* __restrict__ mask,
                           float* __restrict__ out)
{
    int idx = blockIdx.x * blockDim.x + threadIdx.x;
    if (idx >= 2 * 64 * HWF) return;
    int p = idx % HWF;
    int rest = idx / HWF;
    int b = rest % 8; rest /= 8;
    int a = rest % 8;
    int c = rest / 8;
    int y = p / WF, x = p % WF;

    float mv[9], mx = -1e30f;
    #pragma unroll
    for (int t = 0; t < 9; ++t) {
        mv[t] = mask[((size_t)(t * 64 + a * 8 + b)) * HWF + p];
        mx = fmaxf(mx, mv[t]);
    }
    float se = 0.0f;
    #pragma unroll
    for (int t = 0; t < 9; ++t) { mv[t] = __expf(mv[t] - mx); se += mv[t]; }
    float acc = 0.0f;
    #pragma unroll
    for (int t = 0; t < 9; ++t) {
        int yy = y - 1 + t / 3, xx = x - 1 + t % 3;
        float fv = (yy >= 0 && yy < HF && xx >= 0 && xx < WF)
                   ? 8.0f * flow[(size_t)c * HWF + yy * WF + xx] : 0.0f;
        acc += mv[t] * fv;
    }
    out[((size_t)c * (HF * 8) + (y * 8 + a)) * (WF * 8) + (x * 8 + b)] = acc / se;
}

// ---------------------------------------------------------------------------
//                                host side
// ---------------------------------------------------------------------------
struct ConvP { const float* w; const float* b; };
struct AffP  { const float* g = nullptr; const float* b = nullptr; };
struct BlockP { ConvP c1{}, c2{}, down{}; AffP n1, n2, n3; bool hasDown = false; };
struct EncP  { ConvP conv1{}, conv2{}; AffP n0; BlockP blk[6]; bool bn = false; };

static inline int cdiv_(int a, int b) { return (a + b - 1) / b; }

extern "C" void kernel_launch(void* const* d_in, const int* in_sizes, int n_in,
                              void* d_out, int out_size, void* d_ws, size_t ws_size,
                              hipStream_t stream)
{
    if (n_in < 127) return;
    const float* img1 = (const float*)d_in[0];
    const float* img2 = (const float*)d_in[1];
    (void)in_sizes; (void)out_size;

    // ---- parse params (JAX dict-flatten order: cnet, fnet, update) ----
    int pi = 2;
    auto nextp   = [&]() { return (const float*)d_in[pi++]; };
    auto getConv = [&]() { ConvP c; c.w = nextp(); c.b = nextp(); return c; };
    auto getAff  = [&]() { AffP a; a.g = nextp(); a.b = nextp(); return a; };
    auto getEnc  = [&](bool bn) {
        EncP e; e.bn = bn;
        const int hasDown[6] = {0, 0, 1, 0, 1, 0};
        for (int i = 0; i < 6; ++i) {            // 'blocks' sorts first
            BlockP& b = e.blk[i];
            b.c1 = getConv(); b.c2 = getConv();
            b.hasDown = hasDown[i] != 0;
            if (b.hasDown) b.down = getConv();
            if (bn) { b.n1 = getAff(); b.n2 = getAff(); if (b.hasDown) b.n3 = getAff(); }
        }
        e.conv1 = getConv(); e.conv2 = getConv();
        if (bn) e.n0 = getAff();
        return e;
    };
    EncP cnetp = getEnc(true);
    EncP fnetp = getEnc(false);
    ConvP u_conv = getConv(), u_c1 = getConv(), u_c2 = getConv(),
          u_f1 = getConv(), u_f2 = getConv(),
          u_q1 = getConv(), u_q2 = getConv(),
          u_r1 = getConv(), u_r2 = getConv(),
          u_z1 = getConv(), u_z2 = getConv(),
          u_fh1 = getConv(), u_fh2 = getConv(),
          u_m1 = getConv(), u_m2 = getConv();
    const int ITERS = 12;   // reference value; device scalar unreadable in capture

    // ---- workspace bump allocator ----
    size_t wused = 0;
    auto allocB = [&](size_t bytes) {
        void* p = (char*)d_ws + wused;
        wused += (bytes + 255) & ~(size_t)255;
        return p;
    };
    auto allocF = [&](size_t n) { return (float*)allocB(n * sizeof(float)); };

    float* imgcat  = allocF((size_t)2 * 3 * HIN * WIN);
    float* t0      = allocF((size_t)2 * 64 * 184 * 384);
    float* t1      = allocF((size_t)2 * 64 * 184 * 384);
    float* t2      = allocF((size_t)2 * 64 * 184 * 384);
    float* fboth   = allocF((size_t)2 * 256 * HWF);
    _Float16* fhalf = (_Float16*)allocB((size_t)2 * 256 * HWF * sizeof(_Float16));
    float* corr    = allocF((size_t)HWF * HWF);
    float* pyr1    = allocF((size_t)HWF * 23 * 48);
    float* pyr2    = allocF((size_t)HWF * 11 * 24);
    float* pyr3    = allocF((size_t)HWF * 5 * 12);
    float* cnetout = allocF((size_t)256 * HWF);
    float* netbuf  = allocF((size_t)128 * HWF);
    float* hx      = allocF((size_t)384 * HWF);   // [net | inp | motion]
    float* rhx     = allocF((size_t)384 * HWF);   // [r*net | inp | motion]
    float* corrf   = allocF((size_t)324 * HWF);
    float* cbuf    = allocF((size_t)256 * HWF);
    float* cm      = allocF((size_t)256 * HWF);   // [c(192) | f(64)]
    float* fbuf    = allocF((size_t)128 * HWF);
    float* zpre    = allocF((size_t)128 * HWF);
    float* rpre    = allocF((size_t)128 * HWF);
    float* qpre    = allocF((size_t)128 * HWF);
    float* zbuf    = allocF((size_t)128 * HWF);
    float* big     = allocF((size_t)256 * HWF);
    float* maskb   = allocF((size_t)576 * HWF);
    float* dflow   = allocF((size_t)2 * HWF);
    float* coords1 = allocF((size_t)2 * HWF);
    float* flowb   = allocF((size_t)2 * HWF);
    if (wused > ws_size) return;

    // ---- launch helpers ----
    auto convL = [&](const float* in, float* out, int Nimg, int Cin, int Hi, int Wi,
                     ConvP cp, int Cout, int KH, int KW, int s, int pH, int pW,
                     int act, float oscale, int* HoOut = nullptr, int* WoOut = nullptr) {
        int Ho = (Hi + 2 * pH - KH) / s + 1;
        int Wo = (Wi + 2 * pW - KW) / s + 1;
        int M = Nimg * Ho * Wo;
        int nblk = cdiv_(M, 32) * cdiv_(Cout, 32);
        k_conv_wmma<<<nblk, 32, 0, stream>>>(in, cp.w, cp.b, out, Nimg, Cin, Hi, Wi,
                                             Cout, KH, KW, s, pH, pW, Ho, Wo, act, oscale);
        if (HoOut) *HoOut = Ho;
        if (WoOut) *WoOut = Wo;
    };
    auto normL = [&](float* x, int Nimg, int C, int HW, const AffP* a, int relu) {
        k_norm<<<Nimg * C, 256, 0, stream>>>(x, a ? a->g : nullptr,
                                             a ? a->b : nullptr, C, HW, relu);
    };
    auto copyL = [&](const float* s, float* d, int n) {
        k_copy<<<cdiv_(n, 256), 256, 0, stream>>>(s, d, n);
    };

    auto runEnc = [&](const float* x0, int Nimg, const EncP& ep, float* out256) {
        int H, W, C = 64;
        convL(x0, t0, Nimg, 3, HIN, WIN, ep.conv1, 64, 7, 7, 2, 3, 3, 0, 1.0f, &H, &W);
        normL(t0, Nimg, 64, H * W, ep.bn ? &ep.n0 : nullptr, 1);
        const int co[6] = {64, 64, 96, 96, 128, 128};
        const int st[6] = {1, 1, 2, 1, 2, 1};
        for (int i = 0; i < 6; ++i) {
            const BlockP& b = ep.blk[i];
            int Ho, Wo;
            convL(t0, t1, Nimg, C, H, W, b.c1, co[i], 3, 3, st[i], 1, 1, 0, 1.0f, &Ho, &Wo);
            normL(t1, Nimg, co[i], Ho * Wo, ep.bn ? &b.n1 : nullptr, 1);
            convL(t1, t2, Nimg, co[i], Ho, Wo, b.c2, co[i], 3, 3, 1, 1, 1, 0, 1.0f);
            normL(t2, Nimg, co[i], Ho * Wo, ep.bn ? &b.n2 : nullptr, 1);
            const float* xres = t0;
            if (b.hasDown) {
                convL(t0, t1, Nimg, C, H, W, b.down, co[i], 1, 1, st[i], 0, 0, 0, 1.0f);
                normL(t1, Nimg, co[i], Ho * Wo, ep.bn ? &b.n3 : nullptr, 0);
                xres = t1;
            }
            int n = Nimg * co[i] * Ho * Wo;
            k_add_relu<<<cdiv_(n, 256), 256, 0, stream>>>(xres, t2, t0, n);
            C = co[i]; H = Ho; W = Wo;
        }
        convL(t0, out256, Nimg, C, H, W, ep.conv2, 256, 1, 1, 1, 0, 0, 0, 1.0f);
    };

    // ================== pipeline ==================
    const int NIMGELEM = 3 * HIN * WIN;
    k_scale2m1<<<cdiv_(NIMGELEM, 256), 256, 0, stream>>>(img1, imgcat, NIMGELEM);
    k_scale2m1<<<cdiv_(NIMGELEM, 256), 256, 0, stream>>>(img2, imgcat + NIMGELEM, NIMGELEM);

    // fnet on both images (instance norm)
    runEnc(imgcat, 2, fnetp, fboth);

    // correlation volume (f16 WMMA GEMM on transposed [pos][chan] layout)
    k_cvt_t<<<cdiv_(2 * 256 * HWF, 256), 256, 0, stream>>>(fboth, fhalf, 2);
    k_corr_wmma<<<(HWF / 32) * (HWF / 32), 32, 0, stream>>>(
        fhalf, fhalf + (size_t)256 * HWF, corr, 256, HWF, 1.0f / 16.0f);
    k_avgpool<<<cdiv_(HWF * 23 * 48, 256), 256, 0, stream>>>(corr, pyr1, HWF, 46, 96);
    k_avgpool<<<cdiv_(HWF * 11 * 24, 256), 256, 0, stream>>>(pyr1, pyr2, HWF, 23, 48);
    k_avgpool<<<cdiv_(HWF * 5 * 12, 256), 256, 0, stream>>>(pyr2, pyr3, HWF, 11, 24);

    // cnet on image1 (batch norm, training stats)
    runEnc(imgcat, 1, cnetp, cnetout);
    k_split_cnet<<<cdiv_(128 * HWF, 256), 256, 0, stream>>>(cnetout, netbuf, hx, rhx, 128 * HWF);
    k_init_coords<<<cdiv_(HWF, 256), 256, 0, stream>>>(coords1);

    const int NCH = 128 * HWF;
    for (int it = 0; it < ITERS; ++it) {
        k_corr_lookup<<<cdiv_(324 * HWF, 256), 256, 0, stream>>>(
            corr, pyr1, pyr2, pyr3, coords1, corrf);
        k_flow<<<cdiv_(HWF, 256), 256, 0, stream>>>(coords1, flowb);

        // motion encoder
        convL(corrf, cbuf, 1, 324, HF, WF, u_c1, 256, 1, 1, 1, 0, 0, 1, 1.0f);
        convL(cbuf, cm, 1, 256, HF, WF, u_c2, 192, 3, 3, 1, 1, 1, 1, 1.0f);
        convL(flowb, fbuf, 1, 2, HF, WF, u_f1, 128, 7, 7, 1, 3, 3, 1, 1.0f);
        convL(fbuf, cm + (size_t)192 * HWF, 1, 128, HF, WF, u_f2, 64, 3, 3, 1, 1, 1, 1, 1.0f);
        convL(cm, hx + (size_t)256 * HWF, 1, 256, HF, WF, u_conv, 126, 3, 3, 1, 1, 1, 1, 1.0f);
        copyL(flowb, hx + (size_t)382 * HWF, 2 * HWF);          // motion = [m | flow]
        copyL(hx + (size_t)256 * HWF, rhx + (size_t)256 * HWF, NCH);

        // GRU half 1: 1x5 convs, pad (0,2)
        copyL(netbuf, hx, NCH);
        convL(hx, zpre, 1, 384, HF, WF, u_z1, 128, 1, 5, 1, 0, 2, 0, 1.0f);
        convL(hx, rpre, 1, 384, HF, WF, u_r1, 128, 1, 5, 1, 0, 2, 0, 1.0f);
        k_gru_zr<<<cdiv_(NCH, 256), 256, 0, stream>>>(zpre, rpre, netbuf, zbuf, rhx, NCH);
        convL(rhx, qpre, 1, 384, HF, WF, u_q1, 128, 1, 5, 1, 0, 2, 0, 1.0f);
        k_gru_out<<<cdiv_(NCH, 256), 256, 0, stream>>>(qpre, zbuf, netbuf, NCH);

        // GRU half 2: 5x1 convs, pad (2,0)
        copyL(netbuf, hx, NCH);
        convL(hx, zpre, 1, 384, HF, WF, u_z2, 128, 5, 1, 1, 2, 0, 0, 1.0f);
        convL(hx, rpre, 1, 384, HF, WF, u_r2, 128, 5, 1, 1, 2, 0, 0, 1.0f);
        k_gru_zr<<<cdiv_(NCH, 256), 256, 0, stream>>>(zpre, rpre, netbuf, zbuf, rhx, NCH);
        convL(rhx, qpre, 1, 384, HF, WF, u_q2, 128, 5, 1, 1, 2, 0, 0, 1.0f);
        k_gru_out<<<cdiv_(NCH, 256), 256, 0, stream>>>(qpre, zbuf, netbuf, NCH);

        // flow head + mask head
        convL(netbuf, big, 1, 128, HF, WF, u_fh1, 256, 3, 3, 1, 1, 1, 1, 1.0f);
        convL(big, dflow, 1, 256, HF, WF, u_fh2, 2, 3, 3, 1, 1, 1, 0, 1.0f);
        convL(netbuf, big, 1, 128, HF, WF, u_m1, 256, 3, 3, 1, 1, 1, 1, 1.0f);
        convL(big, maskb, 1, 256, HF, WF, u_m2, 576, 1, 1, 1, 0, 0, 0, 0.25f);

        k_add_ip<<<cdiv_(2 * HWF, 256), 256, 0, stream>>>(coords1, dflow, 2 * HWF);
    }

    k_flow<<<cdiv_(HWF, 256), 256, 0, stream>>>(coords1, flowb);
    k_upsample<<<cdiv_(2 * 64 * HWF, 256), 256, 0, stream>>>(flowb, maskb, (float*)d_out);
}
